// DualDomainMamba_33998961116009
// MI455X (gfx1250) — matrix operand order repaired and checked by reading, not compiled, not run on hardware
//
#include <hip/hip_runtime.h>
#include <hip/hip_bf16.h>
#include <math.h>

// ---------------------------------------------------------------------------
// DualDomainMamba for MI455X (gfx1250): bf16 WMMA GEMMs + scan/conv kernels.
// ---------------------------------------------------------------------------

typedef __attribute__((ext_vector_type(16))) __bf16 v16bf;
typedef __attribute__((ext_vector_type(8)))  float  v8f;

#define BM 128     // block tile M (per-wave 32 rows, 4 waves)
#define BN 64      // block tile N
#define BK 32      // K step (one WMMA K)
#define LDA_S 40   // LDS row stride for A tile, elements (20 u32)
#define LDB_S 42   // LDS row stride for B^T tile, elements (21 u32)

// Problem constants (match reference)
#define BSZ     4
#define SEQ     2048
#define DMODEL  512
#define DINNER  1024
#define DSTATE  16
#define DTRANK  32
#define NROWS   (BSZ * SEQ)        // 8192

union frag16 { v16bf v; unsigned int u[8]; };

static __device__ inline unsigned int pack_bf16(float a, float b) {
    union { __bf16 h[2]; unsigned int u; } p;
    p.h[0] = (__bf16)a; p.h[1] = (__bf16)b;
    return p.u;
}

// ---------------------------------------------------------------------------
// Generic fp32-in / fp32-out GEMM with bf16 WMMA core.
//   C[M,N] = A[M,K] * B[K,N]   (row-major, arbitrary leading dims)
//   epi: 0 = none, 1 = + bias[col], 2 = softplus(acc + bias[col])
// Requires M%128==0, N%64==0, K%32==0 (true for all launches below).
// Per-wave: 32x64 output = 8 WMMA accumulators; 8 independent
// v_wmma_f32_16x16x32_bf16 per K-step (hides bf16 WMMA hazard nops).
// ---------------------------------------------------------------------------
__global__ __launch_bounds__(128) void gemm_wmma_bf16(
    const float* __restrict__ A, const float* __restrict__ B,
    float* __restrict__ C, int M, int N, int K,
    int lda, int ldb, int ldc, const float* __restrict__ bias, int epi)
{
    __shared__ __bf16 As[BM * LDA_S];      // [row][k], k packed in pairs
    __shared__ __bf16 Bs[BN * LDB_S];      // transposed: [col][k]

    unsigned int* AsU = (unsigned int*)As; // row stride 20 u32
    unsigned int* BsU = (unsigned int*)Bs; // row stride 21 u32

    const int tid  = threadIdx.x;
    const int lane = tid & 31;
    const int wave = tid >> 5;             // 0..3
    const int m0 = blockIdx.y * BM;
    const int n0 = blockIdx.x * BN;

    // B staging: thread covers k-pair (kp,kp+1) x 8 cols
    const int kp = (tid >> 3) << 1;        // 0,2,..,30
    const int cg = (tid & 7) << 3;         // 0,8,..,56

    v8f acc[8] = {};                       // [i][j] = acc[i*4+j], i:M-sub, j:N-sub

    const int wm = wave << 5;              // wave's 32-row stripe
    const int Ml = lane & 15;              // M (A) / N (B) index within frag
    const int kg = lane >> 4;              // K-group select

    for (int k0 = 0; k0 < K; k0 += BK) {
        // ---- stage A tile: thread t loads row t (32 floats), packs 16 u32 ----
        {
            const float* gA = A + (size_t)(m0 + tid) * lda + k0;
            unsigned int* s = &AsU[tid * (LDA_S / 2)];
            #pragma unroll
            for (int c = 0; c < 16; ++c)
                s[c] = pack_bf16(gA[2 * c], gA[2 * c + 1]);
        }
        // ---- stage B tile transposed: pack (k,k+1) pairs per column ----
        {
            const float* gB0 = B + (size_t)(k0 + kp) * ldb + n0 + cg;
            const float* gB1 = gB0 + ldb;
            #pragma unroll
            for (int c = 0; c < 8; ++c)
                BsU[(cg + c) * (LDB_S / 2) + (kp >> 1)] = pack_bf16(gB0[c], gB1[c]);
        }
        if (k0 + BK < K) {   // prefetch next K-tile (global_prefetch_b8)
            __builtin_prefetch(A + (size_t)(m0 + tid) * lda + (k0 + BK), 0, 1);
            __builtin_prefetch(B + (size_t)(k0 + BK + kp) * ldb + n0 + cg, 0, 1);
        }
        __syncthreads();

        // ---- A fragments (16x32 bf16), ISA layout:
        // lanes 0-15: M=lane, K={0..7,16..23}; lanes 16-31: K={8..15,24..31}
        frag16 af[2];
        #pragma unroll
        for (int i = 0; i < 2; ++i) {
            const unsigned int* arow = &AsU[(wm + (i << 4) + Ml) * (LDA_S / 2)];
            #pragma unroll
            for (int p = 0; p < 8; ++p) {
                const int uidx = (p < 4) ? ((kg << 2) + p)
                                         : (8 + (kg << 2) + (p - 4));
                af[i].u[p] = arow[uidx];
            }
        }
        // ---- B fragments (32x16): lanes 0-15: K={0..7,16..23}, 16-31: K={8..15,24..31} ----
        frag16 bf[4];
        #pragma unroll
        for (int j = 0; j < 4; ++j) {
            const unsigned int* bcol =
                &BsU[((j << 4) + Ml) * (LDB_S / 2) + (kg << 2)];
            #pragma unroll
            for (int p = 0; p < 8; ++p) bf[j].u[p] = bcol[(p < 4) ? p : (p + 4)];
        }
        // ---- 8 mutually independent WMMAs ----
        #pragma unroll
        for (int i = 0; i < 2; ++i)
            #pragma unroll
            for (int j = 0; j < 4; ++j)
                acc[i * 4 + j] = __builtin_amdgcn_wmma_f32_16x16x32_bf16(
                    false, af[i].v, false, bf[j].v, (short)0, acc[i * 4 + j],
                    false, false);
        __syncthreads();
    }

    // ---- epilogue: C/D layout — VGPR v: row = v + 8*(lane>>4), col = lane&15
    #pragma unroll
    for (int i = 0; i < 2; ++i) {
        const int rbase = m0 + wm + (i << 4) + (kg << 3);
        #pragma unroll
        for (int j = 0; j < 4; ++j) {
            const int col = n0 + (j << 4) + Ml;
            if (col >= N) continue;
            const float bv = (epi != 0) ? bias[col] : 0.0f;
            #pragma unroll
            for (int v = 0; v < 8; ++v) {
                const int row = rbase + v;
                if (row >= M) continue;
                float r = acc[i * 4 + j][v];
                if (epi == 1) { r += bv; }
                else if (epi == 2) { r += bv; r = (r > 20.0f) ? r : log1pf(__expf(r)); }
                C[(size_t)row * ldc + col] = r;
            }
        }
    }
}

// ---------------------------------------------------------------------------
// Causal depthwise conv (width 4) + bias + SiLU.
//   xz  : [B, L, 2*DINNER], channel d of the "xi" half = col d
//   out : [B, L, DINNER]
// ---------------------------------------------------------------------------
__global__ __launch_bounds__(256) void conv_silu_k(
    const float* __restrict__ xz, const float* __restrict__ w,
    const float* __restrict__ bias, float* __restrict__ xi)
{
    const int idx = blockIdx.x * 256 + threadIdx.x;    // < 8192*1024
    const int d = idx & (DINNER - 1);
    const int l = (idx >> 10) & (SEQ - 1);
    const int b = idx >> 21;
    const size_t rb = (size_t)b * SEQ;
    float s = bias[d];
    #pragma unroll
    for (int k = 0; k < 4; ++k) {
        const int ls = l - 3 + k;
        if (ls >= 0) s += w[(d << 2) + k] * xz[(rb + ls) * (2 * DINNER) + d];
    }
    xi[idx] = s / (1.0f + __expf(-s));                 // silu
}

// ---------------------------------------------------------------------------
// Sequential SSM scan: thread-per-(b,d) channel, state h[16] in registers,
// per-step B/C vectors staged in LDS (shared across the 256 channels/block).
//   delta,xi : [B,L,DINNER]; xdbl : [B*L,64] (B=cols 32..47, C=cols 48..63)
// ---------------------------------------------------------------------------
__global__ __launch_bounds__(256) void ssm_scan_k(
    const float* __restrict__ delta, const float* __restrict__ xi,
    const float* __restrict__ xdbl, const float* __restrict__ A_log,
    const float* __restrict__ Dp, float* __restrict__ yout)
{
    __shared__ float sBC[2 * DSTATE];
    const int b = blockIdx.x >> 2;
    const int d = ((blockIdx.x & 3) << 8) + threadIdx.x;

    float a[DSTATE], h[DSTATE];
    #pragma unroll
    for (int n = 0; n < DSTATE; ++n) {
        a[n] = -__expf(A_log[d * DSTATE + n]);
        h[n] = 0.0f;
    }
    const float Dv = Dp[d];

    for (int l = 0; l < SEQ; ++l) {
        const size_t r = (size_t)b * SEQ + l;
        if (threadIdx.x < 2 * DSTATE)
            sBC[threadIdx.x] = xdbl[r * 64 + DTRANK + threadIdx.x];
        __syncthreads();
        const float dt = delta[r * DINNER + d];
        const float u  = xi[r * DINNER + d];
        const float du = dt * u;
        float y = 0.0f;
        #pragma unroll
        for (int n = 0; n < DSTATE; ++n) {
            h[n] = __expf(dt * a[n]) * h[n] + du * sBC[n];
            y += h[n] * sBC[DSTATE + n];
        }
        yout[r * DINNER + d] = y + u * Dv;
        __syncthreads();
    }
}

// ---------------------------------------------------------------------------
// Gate: y *= silu(z), z = xz[row, DINNER + col]
// ---------------------------------------------------------------------------
__global__ __launch_bounds__(256) void gate_k(
    float* __restrict__ y, const float* __restrict__ xz)
{
    const int idx = blockIdx.x * 256 + threadIdx.x;    // < 8192*1024
    const int row = idx >> 10;
    const int col = idx & (DINNER - 1);
    const float z = xz[(size_t)row * (2 * DINNER) + DINNER + col];
    y[idx] *= z / (1.0f + __expf(-z));
}

// ---------------------------------------------------------------------------
// Cosine DFT matrix (ortho-normalized rfft real part), rows k>SEQ/2 zero.
//   dft[k][l] = (k <= 1024) ? cos(2*pi*k*l/2048)/sqrt(2048) : 0
// ---------------------------------------------------------------------------
__global__ __launch_bounds__(256) void dft_gen_k(float* __restrict__ dft)
{
    const int idx = blockIdx.x * 256 + threadIdx.x;    // < 2048*2048
    const int k = idx >> 11;
    const int l = idx & (SEQ - 1);
    const float C0   = 6.283185307179586f / (float)SEQ;
    const float INVS = 0.022097086912079612f;          // 1/sqrt(2048)
    float v = 0.0f;
    if (k <= SEQ / 2) {
        const int ph = (k * l) & (SEQ - 1);            // exact phase reduction
        v = __cosf(C0 * (float)ph) * INVS;
    }
    dft[idx] = v;
}

// ---------------------------------------------------------------------------
// Host-side orchestration
// ---------------------------------------------------------------------------
extern "C" void kernel_launch(void* const* d_in, const int* in_sizes, int n_in,
                              void* d_out, int out_size, void* d_ws, size_t ws_size,
                              hipStream_t stream)
{
    (void)in_sizes; (void)n_in; (void)out_size; (void)ws_size;

    const float* x = (const float*)d_in[0];
    auto T = [&](int i) { return (const float*)d_in[i]; };
    float* out = (float*)d_out;

    // workspace layout (floats)
    const size_t SZ_XZ   = (size_t)NROWS * 2 * DINNER;   // 16.78M
    const size_t SZ_XI   = (size_t)NROWS * DINNER;       //  8.39M
    const size_t SZ_XDBL = (size_t)NROWS * 64;           //  0.52M
    const size_t SZ_XFR  = (size_t)BSZ * SEQ * DMODEL;   //  4.19M

    float* ws    = (float*)d_ws;
    float* xz    = ws;
    float* xi    = xz    + SZ_XZ;
    float* xdbl  = xi    + SZ_XI;
    float* delta = xdbl  + SZ_XDBL;
    float* ybuf  = delta + SZ_XI;
    float* cat   = ybuf  + SZ_XI;     // [8192, 1024]: time | freq halves
    float* xfr   = cat   + SZ_XI;
    float* dft   = xfr   + SZ_XFR;    // [2048, 2048]

    const dim3 blk128(128), blk256(256);
    auto gemm = [&](const float* A, const float* Bm, float* C,
                    int M, int N, int K, int lda, int ldb, int ldc,
                    const float* bias, int epi) {
        dim3 grid(N / BN, M / BM);
        gemm_wmma_bf16<<<grid, blk128, 0, stream>>>(A, Bm, C, M, N, K,
                                                    lda, ldb, ldc, bias, epi);
    };

    auto mamba = [&](const float* inp, int pi, float* outcol) {
        const float* in_w    = T(pi + 0);
        const float* conv_w  = T(pi + 1);
        const float* conv_b  = T(pi + 2);
        const float* xproj_w = T(pi + 3);
        const float* dt_w    = T(pi + 4);
        const float* dt_b    = T(pi + 5);
        const float* A_log   = T(pi + 6);
        const float* Dp      = T(pi + 7);
        const float* out_w   = T(pi + 8);

        // in_proj: [8192,512] @ [512,2048] -> xz
        gemm(inp, in_w, xz, NROWS, 2 * DINNER, DMODEL, DMODEL, 2 * DINNER,
             2 * DINNER, nullptr, 0);
        // depthwise conv + silu -> xi
        conv_silu_k<<<dim3(NROWS * DINNER / 256), blk256, 0, stream>>>(
            xz, conv_w, conv_b, xi);
        // x_proj: [8192,1024] @ [1024,64] -> xdbl
        gemm(xi, xproj_w, xdbl, NROWS, 64, DINNER, DINNER, 64, 64, nullptr, 0);
        // dt proj + softplus(·+dt_b): [8192,32] @ [32,1024] -> delta
        gemm(xdbl, dt_w, delta, NROWS, DINNER, DTRANK, 64, DINNER, DINNER,
             dt_b, 2);
        // sequential selective scan -> ybuf
        ssm_scan_k<<<dim3(BSZ * (DINNER / 256)), blk256, 0, stream>>>(
            delta, xi, xdbl, A_log, Dp, ybuf);
        // gate with silu(z)
        gate_k<<<dim3(NROWS * DINNER / 256), blk256, 0, stream>>>(ybuf, xz);
        // out_proj into the concat buffer half (ldc = 1024)
        gemm(ybuf, out_w, outcol, NROWS, DMODEL, DINNER, DINNER, DMODEL,
             2 * DMODEL, nullptr, 0);
    };

    // ---- time-domain branch -> cat[:, :512]
    mamba(x, 1, cat);

    // ---- rfft real part (ortho) as cosine-DFT GEMM -> xfr
    dft_gen_k<<<dim3(SEQ * SEQ / 256), blk256, 0, stream>>>(dft);
    for (int b = 0; b < BSZ; ++b) {
        gemm(dft, x + (size_t)b * SEQ * DMODEL, xfr + (size_t)b * SEQ * DMODEL,
             SEQ, DMODEL, SEQ, SEQ, DMODEL, DMODEL, nullptr, 0);
    }

    // ---- frequency-domain branch -> cat[:, 512:]
    mamba(xfr, 10, cat + DMODEL);

    // ---- fusion: [8192,1024] @ [1024,512] + bias -> out
    gemm(cat, T(19), out, NROWS, DMODEL, 2 * DMODEL, 2 * DMODEL, DMODEL,
         DMODEL, T(20), 1);
}
